// Qwen2MoeDecoderLayer_7395933684226
// MI455X (gfx1250) — compile-verified, matrix operand order checked by reading, not run on hardware
//
#include <hip/hip_runtime.h>
#include <math.h>

// ---------------- problem constants ----------------
constexpr int S  = 1024;
constexpr int H  = 2048;
constexpr int NH = 16;
constexpr int HD = 128;
constexpr int E  = 16;
constexpr int IM = 1408;
constexpr int IS = 5632;

typedef __bf16 bf16;
typedef __attribute__((ext_vector_type(16))) __bf16 v16bf;
typedef __attribute__((ext_vector_type(8)))  float  v8f;

__device__ __forceinline__ bf16 f2bf(float f) {
  unsigned u = __builtin_bit_cast(unsigned, f);
  u += 0x7FFFu + ((u >> 16) & 1u);              // round to nearest even
  unsigned short s = (unsigned short)(u >> 16);
  return __builtin_bit_cast(bf16, s);
}

// packed f32x2 -> bf16x2 (one dword): round-half-up + single v_perm_b32 pack
__device__ __forceinline__ unsigned f2bf_pk(float a, float b) {
  unsigned ua = __builtin_bit_cast(unsigned, a) + 0x8000u;
  unsigned ub = __builtin_bit_cast(unsigned, b) + 0x8000u;
  // dst = { ub[31:16], ua[31:16] }  (a -> low half, b -> high half)
  return __builtin_amdgcn_perm(ub, ua, 0x07060302u);
}

// ---- CDNA5 async global->LDS copy (guarded; falls back to sync copy) ----
#if defined(__AMDGCN__) && __has_builtin(__builtin_amdgcn_global_load_async_to_lds_b128)
#define ASYNC_LDS 1
typedef int v4i_t __attribute__((vector_size(16)));
typedef __attribute__((address_space(1))) v4i_t* gbl_v4i_ptr;
typedef __attribute__((address_space(3))) v4i_t* lds_v4i_ptr;
__device__ __forceinline__ void async_cp16(void* l, const void* g) {
  __builtin_amdgcn_global_load_async_to_lds_b128(
      (gbl_v4i_ptr)g, (lds_v4i_ptr)l, 0, 0);
}
__device__ __forceinline__ void async_wait() {
  asm volatile("s_wait_asynccnt 0" ::: "memory");
}
#else
#define ASYNC_LDS 0
__device__ __forceinline__ void async_wait() {}
#endif

// ---------------- rmsnorm: one block per token ----------------
__global__ __launch_bounds__(256) void rmsnorm_kernel(
    const float* __restrict__ x, const float* __restrict__ w,
    float* __restrict__ of, bf16* __restrict__ ob) {
  int s = blockIdx.x, t = threadIdx.x;
  const float* xr = x + (size_t)s * H;
  float ss = 0.f;
  for (int i = t; i < H; i += 256) { float v = xr[i]; ss += v * v; }
  __shared__ float red[256];
  red[t] = ss; __syncthreads();
  for (int o = 128; o > 0; o >>= 1) { if (t < o) red[t] += red[t + o]; __syncthreads(); }
  float scale = rsqrtf(red[0] / (float)H + 1e-6f);
  for (int i = t; i < H; i += 256) {
    float v = xr[i] * scale * w[i];
    of[(size_t)s * H + i] = v;
    ob[(size_t)s * H + i] = f2bf(v);
  }
}

// ---------------- generic WMMA GEMM ----------------
// C[M,N](f32) = A[M,K](bf16) @ W[K,N](f32, converted to bf16 while staging)
// Block 256 (8 waves, 4Mx2N), block tile BM=128 BN=128 BK=32, wave tile 32x64.
// Double-buffered LDS, software-pipelined global loads; B is stored
// transposed in LDS so both fragment types load as 2x ds_load_b128.
__global__ __launch_bounds__(256) void gemm_bf16_kernel(
    const bf16* __restrict__ A, const float* __restrict__ W,
    float* __restrict__ C, const float* __restrict__ bias,
    const float* __restrict__ rowScale, int rsStride,
    int M, int N, int Kd, int accumulate) {
  constexpr int BK  = 32;
  constexpr int LDA = 40;   // padded row stride (bf16) to spread LDS banks
  constexpr int LDB = 40;
  __shared__ bf16 lA [2][128 * LDA];   // [buf][m][k]
  __shared__ bf16 lBt[2][128 * LDB];   // [buf][n][k]  (transposed)

  const int t = threadIdx.x;
  const int lane  = t & 31;
  const int wave  = t >> 5;
  const int wm    = wave >> 1, wn = wave & 1;
  const int m0    = blockIdx.y * 128;
  const int n0    = blockIdx.x * 128;
  const int lmod  = lane & 15;
  const int lhalf = lane >> 4;
  const int kb    = lhalf * 8;

  // staging coordinates
  const int nn = t & 127;        // B: column of tile this thread owns
  const int kh = t >> 7;         // B: which 16-k half

  v8f acc[2][4];
  #pragma unroll
  for (int mi = 0; mi < 2; ++mi)
    #pragma unroll
    for (int nt = 0; nt < 4; ++nt)
      #pragma unroll
      for (int j = 0; j < 8; ++j) acc[mi][nt][j] = 0.f;

#if !ASYNC_LDS
  uint4 aregs[2];
#endif
  float bregs[16];

  auto issueA = [&](int buf, int k0) {
    #pragma unroll
    for (int u = 0; u < 2; ++u) {
      int c = t * 2 + u, row = c >> 2, ch = c & 3;   // 128 rows x 4 16B-chunks
      const void* src = A + (size_t)(m0 + row) * Kd + k0 + ch * 8;
#if ASYNC_LDS
      async_cp16(&lA[buf][row * LDA + ch * 8], src);
#else
      aregs[u] = *(const uint4*)src;
#endif
    }
  };
  auto publishA = [&](int buf) {
#if !ASYNC_LDS
    #pragma unroll
    for (int u = 0; u < 2; ++u) {
      int c = t * 2 + u, row = c >> 2, ch = c & 3;
      *(uint4*)&lA[buf][row * LDA + ch * 8] = aregs[u];
    }
#else
    (void)buf;
#endif
  };
  auto issueB = [&](int k0) {
    #pragma unroll
    for (int j = 0; j < 16; ++j)
      bregs[j] = W[(size_t)(k0 + kh * 16 + j) * N + n0 + nn];  // lane-coalesced
  };
  auto publishB = [&](int buf) {
    alignas(16) unsigned tmp[8];
    #pragma unroll
    for (int j = 0; j < 8; ++j) tmp[j] = f2bf_pk(bregs[2 * j], bregs[2 * j + 1]);
    *(uint4*)&lBt[buf][nn * LDB + kh * 16]     = *(uint4*)&tmp[0];
    *(uint4*)&lBt[buf][nn * LDB + kh * 16 + 8] = *(uint4*)&tmp[4];
  };

  issueA(0, 0);
  issueB(0);
  int cur = 0;
  for (int k0 = 0; k0 < Kd; k0 += BK, cur ^= 1) {
    publishA(cur);
    publishB(cur);
    async_wait();
    __syncthreads();
    if (k0 + BK < Kd) { issueA(cur ^ 1, k0 + BK); issueB(k0 + BK); }

    // fragments: ISA layout -> two contiguous 8-halfword chunks each
    v16bf afr[2];
    #pragma unroll
    for (int mi = 0; mi < 2; ++mi) {
      const bf16* ap = &lA[cur][(wm * 32 + mi * 16 + lmod) * LDA + kb];
      #pragma unroll
      for (int j = 0; j < 8; ++j) { afr[mi][j] = ap[j]; afr[mi][j + 8] = ap[16 + j]; }
    }
    #pragma unroll
    for (int nt = 0; nt < 4; ++nt) {
      v16bf bv;
      const bf16* bp = &lBt[cur][(wn * 64 + nt * 16 + lmod) * LDB + kb];
      #pragma unroll
      for (int j = 0; j < 8; ++j) { bv[j] = bp[j]; bv[j + 8] = bp[16 + j]; }
      acc[0][nt] = __builtin_amdgcn_wmma_f32_16x16x32_bf16(
          false, afr[0], false, bv, (short)0, acc[0][nt], false, false);
      acc[1][nt] = __builtin_amdgcn_wmma_f32_16x16x32_bf16(
          false, afr[1], false, bv, (short)0, acc[1][nt], false, false);
    }
  }

  // epilogue: C layout lane=n, vgpr i -> m = i + 8*lhalf
  #pragma unroll
  for (int mi = 0; mi < 2; ++mi) {
    #pragma unroll
    for (int nt = 0; nt < 4; ++nt) {
      int n = n0 + wn * 64 + nt * 16 + lmod;
      float bvv = bias ? bias[n] : 0.f;
      #pragma unroll
      for (int i = 0; i < 8; ++i) {
        int m = m0 + wm * 32 + mi * 16 + i + lhalf * 8;
        float v = acc[mi][nt][i] + bvv;
        if (rowScale) v *= rowScale[(size_t)m * rsStride];
        float* dst = &C[(size_t)m * N + n];
        if (accumulate) *dst += v; else *dst = v;
      }
    }
  }
}

// ---------------- RoPE (neox) on Q and K, write bf16 ----------------
__global__ __launch_bounds__(256) void rope_kernel(
    const int* __restrict__ pos,
    const float* __restrict__ qf, const float* __restrict__ kf,
    bf16* __restrict__ qo, bf16* __restrict__ ko) {
  int idx = blockIdx.x * 256 + threadIdx.x;        // S*NH*64
  if (idx >= S * NH * 64) return;
  int d = idx & 63;
  int hID = (idx >> 6) & (NH - 1);
  int s = idx >> 10;
  float p = (float)pos[s];
  float invf = powf(1000000.0f, -((float)(2 * d)) / (float)HD);
  float c, sn;
  sincosf(p * invf, &c, &sn);
  size_t base = (size_t)s * (NH * HD) + hID * HD + d;
  float x1 = qf[base], x2 = qf[base + 64];
  qo[base]      = f2bf(x1 * c - x2 * sn);
  qo[base + 64] = f2bf(x2 * c + x1 * sn);
  x1 = kf[base]; x2 = kf[base + 64];
  ko[base]      = f2bf(x1 * c - x2 * sn);
  ko[base + 64] = f2bf(x2 * c + x1 * sn);
}

__global__ __launch_bounds__(256) void f32_to_bf16_kernel(
    const float* __restrict__ in, bf16* __restrict__ out, int n) {
  int i = (blockIdx.x * 256 + threadIdx.x) * 2;
  if (i < n) {
    *(unsigned*)(out + i) = f2bf_pk(in[i], in[i + 1]);
  }
}

// ---------------- flash attention, WMMA for QK^T and PV ----------------
// grid (S/64, NH), block 128 (4 waves); each wave owns a 16-row Q tile.
__global__ __launch_bounds__(128) void attn_kernel(
    const bf16* __restrict__ Q, const bf16* __restrict__ Kk,
    const bf16* __restrict__ V, bf16* __restrict__ O) {
  __shared__ bf16 lQ[64 * 128];
  __shared__ bf16 lK[32 * 128];
  __shared__ bf16 lVt[128 * 32];       // V transposed: [d][key]
  __shared__ bf16 lP[4 * 16 * 32];     // per-wave P staging

  const int t = threadIdx.x;
  const int lane = t & 31, wave = t >> 5;
  const int lmod = lane & 15, lhalf = lane >> 4, kb = lhalf * 8;
  const int qbase = blockIdx.x * 64;
  const int h = blockIdx.y;

  for (int c = t; c < 512; c += 128) {           // 64 rows x 8 chunks(16B)
    int row = c >> 3, cc = c & 7;
    *(uint4*)&lQ[row * 128 + cc * 8] =
        *(const uint4*)(Q + (size_t)(qbase + row) * (NH * HD) + h * HD + cc * 8);
  }
  __syncthreads();

  v16bf qfr[4];
  #pragma unroll
  for (int f = 0; f < 4; ++f) {
    const bf16* qp = &lQ[(wave * 16 + lmod) * 128 + f * 32 + kb];
    #pragma unroll
    for (int j = 0; j < 8; ++j) { qfr[f][j] = qp[j]; qfr[f][j + 8] = qp[16 + j]; }
  }

  float mrow[8], lrow[8];
  v8f acc[8];
  #pragma unroll
  for (int i = 0; i < 8; ++i) {
    mrow[i] = -1e30f; lrow[i] = 0.f;
    #pragma unroll
    for (int j = 0; j < 8; ++j) acc[i][j] = 0.f;
  }

  const int kend = qbase + 64;
  for (int kblk = 0; kblk < kend; kblk += 32) {
    __syncthreads();                              // prev-iter readers done
    for (int c = t; c < 256; c += 128) {          // 32 rows x 8 chunks
      int row = c >> 3, cc = c & 7;
      *(uint4*)&lK[row * 128 + cc * 8] =
          *(const uint4*)(Kk + (size_t)(kblk + row) * (NH * HD) + h * HD + cc * 8);
      alignas(16) bf16 tmp[8];
      *(uint4*)tmp =
          *(const uint4*)(V + (size_t)(kblk + row) * (NH * HD) + h * HD + cc * 8);
      #pragma unroll
      for (int j = 0; j < 8; ++j) lVt[(cc * 8 + j) * 32 + row] = tmp[j];
    }
    __syncthreads();

    // scores: two 16-key tiles, contraction over d=128 (4 x K32 WMMA)
    v8f sc[2];
    #pragma unroll
    for (int half = 0; half < 2; ++half) {
      #pragma unroll
      for (int j = 0; j < 8; ++j) sc[half][j] = 0.f;
      #pragma unroll
      for (int f = 0; f < 4; ++f) {
        v16bf bv;
        const bf16* kp = &lK[(half * 16 + lmod) * 128 + f * 32 + kb];
        #pragma unroll
        for (int j = 0; j < 8; ++j) { bv[j] = kp[j]; bv[j + 8] = kp[16 + j]; }
        sc[half] = __builtin_amdgcn_wmma_f32_16x16x32_bf16(
            false, qfr[f], false, bv, (short)0, sc[half], false, false);
      }
    }
    // scale + causal mask
    const float ss = 0.08838834764831845f;        // 1/sqrt(128)
    #pragma unroll
    for (int half = 0; half < 2; ++half) {
      int kg = kblk + half * 16 + lmod;
      #pragma unroll
      for (int i = 0; i < 8; ++i) {
        int qg = qbase + wave * 16 + i + lhalf * 8;
        float v = sc[half][i] * ss;
        sc[half][i] = (kg <= qg) ? v : -1e30f;
      }
    }
    // online softmax (row m lives in one 16-lane half -> xor masks 1,2,4,8)
    float pr0[8], pr1[8];
    #pragma unroll
    for (int i = 0; i < 8; ++i) {
      float r = fmaxf(sc[0][i], sc[1][i]);
      #pragma unroll
      for (int mo = 1; mo < 16; mo <<= 1) r = fmaxf(r, __shfl_xor(r, mo, 32));
      float mnew = fmaxf(mrow[i], r);
      float psc = __expf(mrow[i] - mnew);
      float p0 = __expf(sc[0][i] - mnew);
      float p1 = __expf(sc[1][i] - mnew);
      float rs = p0 + p1;
      #pragma unroll
      for (int mo = 1; mo < 16; mo <<= 1) rs += __shfl_xor(rs, mo, 32);
      lrow[i] = lrow[i] * psc + rs;
      mrow[i] = mnew;
      pr0[i] = p0; pr1[i] = p1;
      #pragma unroll
      for (int d = 0; d < 8; ++d) acc[d][i] *= psc;
    }
    // stage P (C layout -> LDS), wave-private region
    bf16* pst = &lP[wave * 16 * 32];
    #pragma unroll
    for (int i = 0; i < 8; ++i) {
      int m = i + lhalf * 8;
      pst[m * 32 + lmod]      = f2bf(pr0[i]);
      pst[m * 32 + 16 + lmod] = f2bf(pr1[i]);
    }
    asm volatile("s_wait_dscnt 0" ::: "memory");  // wave-internal LDS RAW

    v16bf pf;
    const bf16* pp = &pst[lmod * 32 + kb];
    #pragma unroll
    for (int j = 0; j < 8; ++j) { pf[j] = pp[j]; pf[j + 8] = pp[16 + j]; }
    #pragma unroll
    for (int d = 0; d < 8; ++d) {
      v16bf vf;
      const bf16* vp = &lVt[(d * 16 + lmod) * 32 + kb];
      #pragma unroll
      for (int j = 0; j < 8; ++j) { vf[j] = vp[j]; vf[j + 8] = vp[16 + j]; }
      acc[d] = __builtin_amdgcn_wmma_f32_16x16x32_bf16(
          false, pf, false, vf, (short)0, acc[d], false, false);
    }
  }
  #pragma unroll
  for (int d = 0; d < 8; ++d) {
    int n = h * HD + d * 16 + lmod;
    #pragma unroll
    for (int i = 0; i < 8; ++i) {
      int m = qbase + wave * 16 + i + lhalf * 8;
      O[(size_t)m * (NH * HD) + n] = f2bf(acc[d][i] / lrow[i]);
    }
  }
}

// ---------------- router: softmax -> top4 -> renorm, plus sigmoid gate ----------------
__global__ __launch_bounds__(256) void router_kernel(
    const float* __restrict__ h2, const float* __restrict__ rw,
    const float* __restrict__ wsg, float* __restrict__ combine,
    float* __restrict__ sgate) {
  int s = blockIdx.x, t = threadIdx.x;
  float p[17];
  #pragma unroll
  for (int e = 0; e < 17; ++e) p[e] = 0.f;
  for (int hh = t; hh < H; hh += 256) {
    float hv = h2[(size_t)s * H + hh];
    const float* r = rw + (size_t)hh * E;
    #pragma unroll
    for (int e = 0; e < E; ++e) p[e] += hv * r[e];
    p[16] += hv * wsg[hh];
  }
  __shared__ float red[256];
  __shared__ float logits[17];
  for (int e = 0; e < 17; ++e) {
    red[t] = p[e]; __syncthreads();
    for (int o = 128; o > 0; o >>= 1) { if (t < o) red[t] += red[t + o]; __syncthreads(); }
    if (t == 0) logits[e] = red[0];
    __syncthreads();
  }
  if (t == 0) {
    float mx = logits[0];
    for (int e = 1; e < E; ++e) mx = fmaxf(mx, logits[e]);
    float pe[E], sum = 0.f;
    for (int e = 0; e < E; ++e) { pe[e] = __expf(logits[e] - mx); sum += pe[e]; }
    for (int e = 0; e < E; ++e) pe[e] /= sum;
    float cw[E]; for (int e = 0; e < E; ++e) cw[e] = 0.f;
    float tsum = 0.f;
    bool used[E]; for (int e = 0; e < E; ++e) used[e] = false;
    for (int k = 0; k < 4; ++k) {
      int best = 0; float bv = -1.f;
      for (int e = 0; e < E; ++e) if (!used[e] && pe[e] > bv) { bv = pe[e]; best = e; }
      used[best] = true; cw[best] = pe[best]; tsum += pe[best];
    }
    for (int e = 0; e < E; ++e) combine[(size_t)s * E + e] = cw[e] / tsum;
    sgate[s] = 1.f / (1.f + __expf(-logits[16]));
  }
}

__global__ __launch_bounds__(256) void silu_mul_kernel(
    const float* __restrict__ g, const float* __restrict__ u,
    bf16* __restrict__ out, int n) {
  int i = (blockIdx.x * 256 + threadIdx.x) * 4;
  if (i < n) {
    float4 gv = *(const float4*)(g + i);
    float4 uv = *(const float4*)(u + i);
    float r0 = (gv.x / (1.f + __expf(-gv.x))) * uv.x;
    float r1 = (gv.y / (1.f + __expf(-gv.y))) * uv.y;
    float r2 = (gv.z / (1.f + __expf(-gv.z))) * uv.z;
    float r3 = (gv.w / (1.f + __expf(-gv.w))) * uv.w;
    uint2 o;
    o.x = f2bf_pk(r0, r1);
    o.y = f2bf_pk(r2, r3);
    *(uint2*)(out + i) = o;
  }
}

__global__ __launch_bounds__(256) void add_kernel(
    float* __restrict__ x, const float* __restrict__ r, int n) {
  int i = blockIdx.x * 256 + threadIdx.x;
  if (i < n) x[i] += r[i];
}

// ---------------- host launch ----------------
extern "C" void kernel_launch(void* const* d_in, const int* in_sizes, int n_in,
                              void* d_out, int out_size, void* d_ws, size_t ws_size,
                              hipStream_t stream) {
  (void)in_sizes; (void)n_in; (void)out_size; (void)ws_size;
  const int*   positions = (const int*)d_in[0];
  const float* hs   = (const float*)d_in[1];
  const float* ln1  = (const float*)d_in[2];
  const float* ln2  = (const float*)d_in[3];
  const float* q_w  = (const float*)d_in[4];
  const float* q_b  = (const float*)d_in[5];
  const float* k_w  = (const float*)d_in[6];
  const float* k_b  = (const float*)d_in[7];
  const float* v_w  = (const float*)d_in[8];
  const float* v_b  = (const float*)d_in[9];
  const float* o_w  = (const float*)d_in[10];
  const float* router_w = (const float*)d_in[11];
  const float* we_gate  = (const float*)d_in[12];
  const float* we_up    = (const float*)d_in[13];
  const float* we_down  = (const float*)d_in[14];
  const float* ws_gate  = (const float*)d_in[15];
  const float* ws_up    = (const float*)d_in[16];
  const float* ws_down  = (const float*)d_in[17];
  const float* wsg      = (const float*)d_in[18];
  float* out = (float*)d_out;

  char* wsp = (char*)d_ws;
  size_t off = 0;
  auto alloc = [&](size_t bytes) -> void* {
    void* p = wsp + off;
    off += (bytes + 255) & ~(size_t)255;
    return p;
  };
  float* h1f = (float*)alloc((size_t)S * H * 4);
  bf16*  h1b = (bf16*) alloc((size_t)S * H * 2);
  float* qf  = (float*)alloc((size_t)S * NH * HD * 4);
  float* kf  = (float*)alloc((size_t)S * NH * HD * 4);
  float* vf  = (float*)alloc((size_t)S * NH * HD * 4);
  bf16*  qbb = (bf16*) alloc((size_t)S * NH * HD * 2);
  bf16*  kbb = (bf16*) alloc((size_t)S * NH * HD * 2);
  bf16*  vbb = (bf16*) alloc((size_t)S * NH * HD * 2);
  bf16*  attnb = (bf16*)alloc((size_t)S * NH * HD * 2);
  float* x2  = (float*)alloc((size_t)S * H * 4);
  float* h2f = (float*)alloc((size_t)S * H * 4);
  bf16*  h2b = (bf16*) alloc((size_t)S * H * 2);
  float* gbuf = (float*)alloc((size_t)S * IS * 4);
  float* ubuf = (float*)alloc((size_t)S * IS * 4);
  bf16*  gub  = (bf16*) alloc((size_t)S * IS * 2);
  float* combine = (float*)alloc((size_t)S * E * 4);
  float* sgate   = (float*)alloc((size_t)S * 4);

  auto gemm = [&](const bf16* A, const float* W, float* C, const float* bias,
                  const float* rs, int rss, int M, int N, int Kd, int acc) {
    dim3 g(N / 128, M / 128);
    gemm_bf16_kernel<<<g, 256, 0, stream>>>(A, W, C, bias, rs, rss, M, N, Kd, acc);
  };

  // --- pre-attention ---
  rmsnorm_kernel<<<S, 256, 0, stream>>>(hs, ln1, h1f, h1b);
  gemm(h1b, q_w, qf, q_b, nullptr, 0, S, NH * HD, H, 0);
  gemm(h1b, k_w, kf, k_b, nullptr, 0, S, NH * HD, H, 0);
  gemm(h1b, v_w, vf, v_b, nullptr, 0, S, NH * HD, H, 0);
  rope_kernel<<<(S * NH * 64) / 256, 256, 0, stream>>>(positions, qf, kf, qbb, kbb);
  f32_to_bf16_kernel<<<(S * NH * HD / 2) / 256, 256, 0, stream>>>(vf, vbb, S * NH * HD);

  // --- attention + o-proj + residual ---
  attn_kernel<<<dim3(S / 64, NH), 128, 0, stream>>>(qbb, kbb, vbb, attnb);
  gemm(attnb, o_w, x2, nullptr, nullptr, 0, S, H, NH * HD, 0);
  add_kernel<<<(S * H) / 256, 256, 0, stream>>>(x2, hs, S * H);

  // --- post-attention norm, router ---
  rmsnorm_kernel<<<S, 256, 0, stream>>>(x2, ln2, h2f, h2b);
  router_kernel<<<S, 256, 0, stream>>>(h2f, router_w, wsg, combine, sgate);

  // --- shared expert: out = sigmoid_gate * (silu(h@wg)*(h@wu)) @ wd ---
  gemm(h2b, ws_gate, gbuf, nullptr, nullptr, 0, S, IS, H, 0);
  gemm(h2b, ws_up,   ubuf, nullptr, nullptr, 0, S, IS, H, 0);
  silu_mul_kernel<<<(S * IS / 4) / 256, 256, 0, stream>>>(gbuf, ubuf, gub, S * IS);
  gemm(gub, ws_down, out, nullptr, sgate, 1, S, H, IS, 0);   // beta=0 initializes out

  // --- experts: out += combine[:,e] * expert_e(h) ---
  for (int e = 0; e < E; ++e) {
    gemm(h2b, we_gate + (size_t)e * H * IM, gbuf, nullptr, nullptr, 0, S, IM, H, 0);
    gemm(h2b, we_up   + (size_t)e * H * IM, ubuf, nullptr, nullptr, 0, S, IM, H, 0);
    silu_mul_kernel<<<(S * IM / 4) / 256, 256, 0, stream>>>(gbuf, ubuf, gub, S * IM);
    gemm(gub, we_down + (size_t)e * IM * H, out, nullptr, combine + e, E, S, H, IM, 1);
  }
}
